// RHLAttentionCouplingDiffeomorphismNet_79147657330875
// MI455X (gfx1250) — compile-verified, matrix operand order checked by reading, not run on hardware
//
#include <hip/hip_runtime.h>
#include <math.h>

typedef __attribute__((ext_vector_type(16))) _Float16 v16h;
typedef __attribute__((ext_vector_type(8)))  _Float16 v8h;
typedef __attribute__((ext_vector_type(8)))  float    v8f;
typedef _Float16 h16;

#define WAVES 2
#define TB    (WAVES * 32)
#define NBT   16          // batch elements per wave (WMMA N)

// ---------------- LDS layout (halves region, then floats region) ----------------
enum : int {
  H_FCWIN = 0,                    // 64x32 (cols 16..31 zero)
  H_FCWH0 = H_FCWIN + 64 * 32,
  H_FCWH1 = H_FCWH0 + 64 * 64,
  H_FCWOUT = H_FCWH1 + 64 * 64,   // 16x64 (rows 8..15 zero)
  H_GWFC  = H_FCWOUT + 16 * 64,   // 16x64 = (Wh0@Win)^T  (row i, col k)
  H_CCWIN = H_GWFC + 16 * 64,     // 64x32 (cols 8..31 zero)
  H_CCWH0 = H_CCWIN + 64 * 32,
  H_CCWH1 = H_CCWH0 + 64 * 64,
  H_CCWOUT = H_CCWH1 + 64 * 64,
  H_GWCC  = H_CCWOUT + 16 * 64,   // 8x64 transposed
  H_FDWIN = H_GWCC + 8 * 64,      // 64x32
  H_FDWH0 = H_FDWIN + 64 * 32,
  H_FDWH1 = H_FDWH0 + 64 * 64,
  H_FDWOUT = H_FDWH1 + 64 * 64,   // 16x64
  H_WAVE  = H_FDWOUT + 16 * 64,   // per-wave: ST(16x64) D1(16x64) D2(16x64)
  H_PW    = 3 * 1024,
  H_TOTAL = H_WAVE + WAVES * H_PW,
};
enum : int {
  F_FCB = 0,                      // bin(64) bh0(64) bh1(64) bout(16, padded)
  F_CCB = F_FCB + 208,
  F_FDB = F_CCB + 208,
  F_WAVE = F_FDB + 208,
  // per-wave floats: EG 2048, FG 2048, TG 2048, NG 1024, VE/VN1/VF1/VN3/VS 128 each, HOUT 256
  F_PW = 2048 * 3 + 1024 + 128 * 5 + 256,   // 8064
  F_TOTAL = F_WAVE + WAVES * F_PW,
};
constexpr size_t LDS_BYTES = (size_t)H_TOTAL * 2 + (size_t)F_TOTAL * 4;  // ~146 KB

// ---------------- fragment helpers ----------------
// Per the CDNA5 ISA 16-bit A layout (16x32): lanes 0-15 hold K {0..7,16..23} of row M=lane,
// lanes 16-31 hold K {8..15,24..31} of row M=lane-16. B mirrors this with N=lane%16,
// provided the staging buffer ST is stored batch-major (row = batch col N, 64 halves wide).
static __device__ __forceinline__ v16h ld_frag(const h16* p) {
  const v8h* q = (const v8h*)p;
  v8h a = q[0];
  v8h b = q[2];
  v16h r;
#pragma unroll
  for (int j = 0; j < 8; ++j) { r[j] = a[j]; r[j + 8] = b[j]; }
  return r;
}

static __device__ __forceinline__ void zero4(v8f a[4]) {
#pragma unroll
  for (int t = 0; t < 4; ++t)
#pragma unroll
    for (int v = 0; v < 8; ++v) a[t][v] = 0.f;
}
static __device__ __forceinline__ void relu4(v8f h[4]) {
#pragma unroll
  for (int t = 0; t < 4; ++t)
#pragma unroll
    for (int v = 0; v < 8; ++v) h[t][v] = fmaxf(h[t][v], 0.f);
}

// OUT(64x16) = W(64xK) @ ST(Kx16) [+ bias];  ksteps = K/32  (call with literal ksteps)
static __device__ __forceinline__ void gemm64(const h16* W, int ldw, int ksteps,
                                              const h16* ST, const float* bias,
                                              v8f acc[4], int lane) {
  const int rr = lane & 15, hi8 = (lane >> 4) * 8;
  zero4(acc);
#pragma unroll 2
  for (int ks = 0; ks < ksteps; ++ks) {
    v16h bf = ld_frag(ST + rr * 64 + ks * 32 + hi8);
#pragma unroll
    for (int t = 0; t < 4; ++t) {
      v16h af = ld_frag(W + (16 * t + rr) * ldw + ks * 32 + hi8);
      acc[t] = __builtin_amdgcn_wmma_f32_16x16x32_f16(false, af, false, bf,
                                                      (short)0, acc[t], false, false);
    }
  }
  if (bias) {
#pragma unroll
    for (int t = 0; t < 4; ++t)
#pragma unroll
      for (int v = 0; v < 8; ++v) acc[t][v] += bias[16 * t + v + hi8];
  }
}

// OUT(16x16) = W(16x64) @ ST(64x16) [+ bias]
static __device__ __forceinline__ void gemm16(const h16* W, const h16* ST,
                                              const float* bias, v8f& acc, int lane) {
  const int rr = lane & 15, hi8 = (lane >> 4) * 8;
#pragma unroll
  for (int v = 0; v < 8; ++v) acc[v] = 0.f;
#pragma unroll
  for (int ks = 0; ks < 2; ++ks) {
    v16h bf = ld_frag(ST + rr * 64 + ks * 32 + hi8);
    v16h af = ld_frag(W + rr * 64 + ks * 32 + hi8);
    acc = __builtin_amdgcn_wmma_f32_16x16x32_f16(false, af, false, bf,
                                                 (short)0, acc, false, false);
  }
  if (bias)
#pragma unroll
    for (int v = 0; v < 8; ++v) acc[v] += bias[v + hi8];
}

// store 64x16 f32 accumulator tiles into batch-major f16 staging (ST[n*64 + feature])
static __device__ __forceinline__ void stage64(const v8f acc[4], h16* ST, int lane) {
  const int n = lane & 15, ro = (lane >> 4) * 8;
#pragma unroll
  for (int t = 0; t < 4; ++t)
#pragma unroll
    for (int v = 0; v < 8; ++v) ST[n * 64 + 16 * t + v + ro] = (h16)acc[t][v];
}
static __device__ __forceinline__ void setD(const v8f p[4], h16* D, int lane) {
  const int n = lane & 15, ro = (lane >> 4) * 8;
#pragma unroll
  for (int t = 0; t < 4; ++t)
#pragma unroll
    for (int v = 0; v < 8; ++v)
      D[n * 64 + 16 * t + v + ro] = (h16)((p[t][v] > 0.f) ? 1.f : 0.f);
}

struct WSet {
  const h16 *Win, *Wh0, *Wh1, *Wout;
  const float *bin, *bh0, *bh1, *bout;
};
struct PW {
  h16 *ST, *D1, *D2;
  float *EG, *FG, *TG, *NG, *VE, *VN1, *VF1, *VN3, *VS, *HOUT;
};

// _diff_jaco forward part: h0=Win@X+b; h1=relu(Wh0@h0+b); h2=relu(Wh1@h1+b); out=Wout@h2+b
// deltas per reference quirk: D1=sign(relu(Wh0@h1+b0)), D2=sign(relu(Wh1@h2+b1))
static __device__ v8f forward_jnet(const WSet& w, h16* ST, h16* D1, h16* D2,
                                   int lane, bool deltas) {
  v8f h[4];
  gemm64(w.Win, 32, 1, ST, w.bin, h, lane);           // h0 (no relu)
  stage64(h, ST, lane);
  gemm64(w.Wh0, 64, 2, ST, w.bh0, h, lane);           // a1
  relu4(h);
  stage64(h, ST, lane);                               // h1
  if (deltas) {
    v8f p[4];
    gemm64(w.Wh0, 64, 2, ST, w.bh0, p, lane);         // Wh0@h1+b0
    setD(p, D1, lane);
  }
  gemm64(w.Wh1, 64, 2, ST, w.bh1, h, lane);           // a2
  relu4(h);
  stage64(h, ST, lane);                               // h2
  if (deltas) {
    v8f p[4];
    gemm64(w.Wh1, 64, 2, ST, w.bh1, p, lane);         // Wh1@h2+b1
    setD(p, D2, lane);
  }
  v8f out;
  gemm16(w.Wout, ST, w.bout, out, lane);
  return out;
}

// Jacobian chain: for each input col i: OUTG[:,i] = Wout @ (D2 .* (Wh1 @ (D1 .* gW[:,i])))
// gWT is stored transposed (NI x 64) so row i is fragment-shaped; the first GEMM's B
// fragment is built entirely in registers (D1 frag * gWT frag, packed f16 mul).
// Wh1/Wout/D1 fragments are loop-invariant and hoisted into VGPRs.
static __device__ void jac_chain(const h16* Wh1, const h16* Wout, const h16* gWT, int NI,
                                 const h16* D1, const h16* D2, h16* ST,
                                 float* OUTG, int lane) {
  const int n = lane & 15, hlf = lane >> 4, ro = hlf * 8, hi8 = ro;
  v16h wa[8], wo[2], d1f[2];
#pragma unroll
  for (int ks = 0; ks < 2; ++ks) {
#pragma unroll
    for (int t = 0; t < 4; ++t)
      wa[ks * 4 + t] = ld_frag(Wh1 + (16 * t + n) * 64 + ks * 32 + hi8);
    wo[ks] = ld_frag(Wout + n * 64 + ks * 32 + hi8);
    d1f[ks] = ld_frag(D1 + n * 64 + ks * 32 + hi8);
  }
  for (int i = 0; i < NI; ++i) {
    v8f t4[4];
    zero4(t4);
#pragma unroll
    for (int ks = 0; ks < 2; ++ks) {
      v16h bf = d1f[ks] * ld_frag(gWT + i * 64 + ks * 32 + hi8);  // v_pk_mul_f16
#pragma unroll
      for (int t = 0; t < 4; ++t)
        t4[t] = __builtin_amdgcn_wmma_f32_16x16x32_f16(false, wa[ks * 4 + t], false, bf,
                                                       (short)0, t4[t], false, false);
    }
    // fused D2 mask + stage
#pragma unroll
    for (int t = 0; t < 4; ++t)
#pragma unroll
      for (int v = 0; v < 8; ++v)
        ST[n * 64 + 16 * t + v + ro] =
            (h16)(t4[t][v] * (float)D2[n * 64 + 16 * t + v + ro]);
    v8f e;
#pragma unroll
    for (int v = 0; v < 8; ++v) e[v] = 0.f;
#pragma unroll
    for (int ks = 0; ks < 2; ++ks) {
      v16h bf = ld_frag(ST + n * 64 + ks * 32 + hi8);
      e = __builtin_amdgcn_wmma_f32_16x16x32_f16(false, wo[ks], false, bf,
                                                 (short)0, e, false, false);
    }
    if (!hlf) {
#pragma unroll
      for (int v = 0; v < 8; ++v) OUTG[(v * NI + i) * 16 + n] = e[v];
    }
  }
}

// _st_network: input already staged in pw.ST (padded to K=32). Leaves s_grad in TG, s_out in VS.
static __device__ void st_network(const WSet& fc, const WSet& cc,
                                  const h16* gWTfc, const h16* gWTcc,
                                  const PW& pw, int lane) {
  const int n = lane & 15, hlf = lane >> 4;
  // E = diff_jaco(x, fc)
  v8f eo = forward_jnet(fc, pw.ST, pw.D1, pw.D2, lane, true);
  if (!hlf) {
#pragma unroll
    for (int v = 0; v < 8; ++v) pw.VE[v * 16 + n] = eo[v];
  }
  jac_chain(fc.Wh1, fc.Wout, gWTfc, 16, pw.D1, pw.D2, pw.ST, pw.EG, lane);

  // N1 = diff_jaco(E_out, fcc)
  for (int k = hlf * 32; k < hlf * 32 + 32; ++k)
    pw.ST[n * 64 + k] = (k < 8) ? (h16)pw.VE[k * 16 + n] : (h16)0.f;
  v8f n1 = forward_jnet(cc, pw.ST, pw.D1, pw.D2, lane, true);
  if (!hlf) {
#pragma unroll
    for (int v = 0; v < 8; ++v) pw.VN1[v * 16 + n] = n1[v];
  }
  jac_chain(cc.Wh1, cc.Wout, gWTcc, 8, pw.D1, pw.D2, pw.ST, pw.NG, lane);

  // NE = N1_grad @ E_grad  -> TG
  for (int o = hlf * 4; o < hlf * 4 + 4; ++o)
    for (int i = 0; i < 16; ++i) {
      float s = 0.f;
#pragma unroll
      for (int j = 0; j < 8; ++j)
        s += pw.NG[(o * 8 + j) * 16 + n] * pw.EG[(j * 16 + i) * 16 + n];
      pw.TG[(o * 16 + i) * 16 + n] = s;
    }
  // F1_grad = E_grad*N1_out + NE*(1+E_out) ; F1_out = E_out*N1_out + N1_out
  for (int o = hlf * 4; o < hlf * 4 + 4; ++o) {
    float e = pw.VE[o * 16 + n], n1o = pw.VN1[o * 16 + n];
    pw.VF1[o * 16 + n] = e * n1o + n1o;
    for (int i = 0; i < 16; ++i) {
      int id = (o * 16 + i) * 16 + n;
      pw.FG[id] = pw.EG[id] * n1o + pw.TG[id] * (1.f + e);
    }
  }
  // N3 = diff_jaco(F1_out, fcc)
  for (int k = hlf * 32; k < hlf * 32 + 32; ++k)
    pw.ST[n * 64 + k] = (k < 8) ? (h16)pw.VF1[k * 16 + n] : (h16)0.f;
  v8f n3 = forward_jnet(cc, pw.ST, pw.D1, pw.D2, lane, true);
  if (!hlf) {
#pragma unroll
    for (int v = 0; v < 8; ++v) pw.VN3[v * 16 + n] = n3[v];
  }
  jac_chain(cc.Wh1, cc.Wout, gWTcc, 8, pw.D1, pw.D2, pw.ST, pw.NG, lane);

  // NF = N3_grad @ F1_grad -> TG
  for (int o = hlf * 4; o < hlf * 4 + 4; ++o)
    for (int i = 0; i < 16; ++i) {
      float s = 0.f;
#pragma unroll
      for (int j = 0; j < 8; ++j)
        s += pw.NG[(o * 8 + j) * 16 + n] * pw.FG[(j * 16 + i) * 16 + n];
      pw.TG[(o * 16 + i) * 16 + n] = s;
    }
  // F2_grad = E_grad + NF*(1+F1_out)  (in place into TG) ; s_out = E_out + F1_out*N3_out
  for (int o = hlf * 4; o < hlf * 4 + 4; ++o) {
    float e = pw.VE[o * 16 + n], f1 = pw.VF1[o * 16 + n], n3o = pw.VN3[o * 16 + n];
    pw.VS[o * 16 + n] = e + f1 * n3o;
    for (int i = 0; i < 16; ++i) {
      int id = (o * 16 + i) * 16 + n;
      pw.TG[id] = pw.EG[id] + pw.TG[id] * (1.f + f1);
    }
  }
}

// ---------------- weight staging helpers ----------------
static __device__ void cvt(h16* dst, const float* src, int R, int C, int Rs, int Cs, int tid) {
  for (int idx = tid; idx < R * C; idx += TB) {
    int r = idx / C, c = idx - r * C;
    dst[idx] = (r < Rs && c < Cs) ? (h16)src[r * Cs + c] : (h16)0.f;
  }
}
static __device__ void cpyb(float* dst, const float* src, int N, int Ns, int tid) {
  for (int i = tid; i < N; i += TB) dst[i] = (i < Ns) ? src[i] : 0.f;
}

__global__ __launch_bounds__(TB) void diffeo_kernel(
    const float* __restrict__ x,
    const float* fcWin_g, const float* fcbin_g, const float* fcWh_g, const float* fcbh_g,
    const float* fcWout_g, const float* fcbout_g,
    const float* ccWin_g, const float* ccbin_g, const float* ccWh_g, const float* ccbh_g,
    const float* ccWout_g, const float* ccbout_g,
    const float* fdWin_g, const float* fdbin_g, const float* fdWh_g, const float* fdbh_g,
    const float* fdWout_g, const float* fdbout_g,
    float* __restrict__ out, int batch) {
  extern __shared__ char smem[];
  h16* HB = (h16*)smem;
  float* FB = (float*)(smem + (size_t)H_TOTAL * sizeof(h16));
  const int tid = threadIdx.x;

  // ---- stage weights (f32 -> f16, padded) ----
  cvt(HB + H_FCWIN, fcWin_g, 64, 32, 64, 16, tid);
  cvt(HB + H_FCWH0, fcWh_g, 64, 64, 64, 64, tid);
  cvt(HB + H_FCWH1, fcWh_g + 4096, 64, 64, 64, 64, tid);
  cvt(HB + H_FCWOUT, fcWout_g, 16, 64, 8, 64, tid);
  cvt(HB + H_CCWIN, ccWin_g, 64, 32, 64, 8, tid);
  cvt(HB + H_CCWH0, ccWh_g, 64, 64, 64, 64, tid);
  cvt(HB + H_CCWH1, ccWh_g + 4096, 64, 64, 64, 64, tid);
  cvt(HB + H_CCWOUT, ccWout_g, 16, 64, 8, 64, tid);
  cvt(HB + H_FDWIN, fdWin_g, 64, 32, 64, 16, tid);
  cvt(HB + H_FDWH0, fdWh_g, 64, 64, 64, 64, tid);
  cvt(HB + H_FDWH1, fdWh_g + 4096, 64, 64, 64, 64, tid);
  cvt(HB + H_FDWOUT, fdWout_g, 16, 64, 16, 64, tid);
  // gWT = (Wh0 @ Win)^T, stored row i (NI) x col f (64), batch-independent
  for (int idx = tid; idx < 16 * 64; idx += TB) {
    int i = idx >> 6, f = idx & 63;
    float s = 0.f;
    for (int k = 0; k < 64; ++k) s += fcWh_g[f * 64 + k] * fcWin_g[k * 16 + i];
    HB[H_GWFC + idx] = (h16)s;
  }
  for (int idx = tid; idx < 8 * 64; idx += TB) {
    int i = idx >> 6, f = idx & 63;
    float s = 0.f;
    for (int k = 0; k < 64; ++k) s += ccWh_g[f * 64 + k] * ccWin_g[k * 8 + i];
    HB[H_GWCC + idx] = (h16)s;
  }
  // biases (f32)
  cpyb(FB + F_FCB, fcbin_g, 64, 64, tid);
  cpyb(FB + F_FCB + 64, fcbh_g, 64, 64, tid);
  cpyb(FB + F_FCB + 128, fcbh_g + 64, 64, 64, tid);
  cpyb(FB + F_FCB + 192, fcbout_g, 16, 8, tid);
  cpyb(FB + F_CCB, ccbin_g, 64, 64, tid);
  cpyb(FB + F_CCB + 64, ccbh_g, 64, 64, tid);
  cpyb(FB + F_CCB + 128, ccbh_g + 64, 64, 64, tid);
  cpyb(FB + F_CCB + 192, ccbout_g, 16, 8, tid);
  cpyb(FB + F_FDB, fdbin_g, 64, 64, tid);
  cpyb(FB + F_FDB + 64, fdbh_g, 64, 64, tid);
  cpyb(FB + F_FDB + 128, fdbh_g + 64, 64, 64, tid);
  cpyb(FB + F_FDB + 192, fdbout_g, 16, 16, tid);
  __syncthreads();

  WSet fc = {HB + H_FCWIN, HB + H_FCWH0, HB + H_FCWH1, HB + H_FCWOUT,
             FB + F_FCB, FB + F_FCB + 64, FB + F_FCB + 128, FB + F_FCB + 192};
  WSet cc = {HB + H_CCWIN, HB + H_CCWH0, HB + H_CCWH1, HB + H_CCWOUT,
             FB + F_CCB, FB + F_CCB + 64, FB + F_CCB + 128, FB + F_CCB + 192};
  WSet fd = {HB + H_FDWIN, HB + H_FDWH0, HB + H_FDWH1, HB + H_FDWOUT,
             FB + F_FDB, FB + F_FDB + 64, FB + F_FDB + 128, FB + F_FDB + 192};

  const int lane = tid & 31, wv = tid >> 5;
  const int n = lane & 15, hlf = lane >> 4;
  PW pw;
  pw.ST = HB + H_WAVE + wv * H_PW;
  pw.D1 = pw.ST + 1024;
  pw.D2 = pw.D1 + 1024;
  float* fw = FB + F_WAVE + wv * F_PW;
  pw.EG = fw;          pw.FG = fw + 2048;  pw.TG = fw + 4096;  pw.NG = fw + 6144;
  pw.VE = fw + 7168;   pw.VN1 = fw + 7296; pw.VF1 = fw + 7424; pw.VN3 = fw + 7552;
  pw.VS = fw + 7680;   pw.HOUT = fw + 7808;

  const int b0 = (blockIdx.x * WAVES + wv) * NBT;
  if (b0 >= batch) return;   // wave-uniform
  const float* xp = x + (size_t)(b0 + n) * 48;
  float* outp = out + (size_t)(b0 + n) * 64;

  // ===================== pass 1: xt =====================
  for (int k = hlf * 32; k < hlf * 32 + 32; ++k)
    pw.ST[n * 64 + k] = (k < 16) ? (h16)xp[k] : (h16)0.f;
  st_network(fc, cc, HB + H_GWFC, HB + H_GWCC, pw, lane);

  float xd[16];
#pragma unroll
  for (int i = 0; i < 16; ++i) xd[i] = xp[16 + i];

  for (int r = hlf * 4; r < hlf * 4 + 4; ++r) {
    float so = pw.VS[r * 16 + n];
    float es = expf(so);
    float x2 = xp[8 + r];
    pw.HOUT[(8 + r) * 16 + n] = es * x2 + so;          // y2
    float sc1 = es * x2 + 1.f;                          // scale + 1
    float s = 0.f;
#pragma unroll
    for (int i = 0; i < 16; ++i) s += pw.TG[(r * 16 + i) * 16 + n] * xd[i];
    outp[16 + 8 + r] = sc1 * s + es * xd[8 + r];        // h_dot rows 8..15
  }
  if (!hlf) {
#pragma unroll
    for (int c = 0; c < 8; ++c) {
      pw.HOUT[c * 16 + n] = xp[c];                      // x1
      outp[16 + c] = xd[c];                             // h_dot rows 0..7
    }
  }
#pragma unroll
  for (int j = 0; j < 8; ++j) {                         // h_out
    int f = hlf * 8 + j;
    outp[f] = pw.HOUT[f * 16 + n];
  }

  // ===================== fd MLP -> x_hat =====================
  for (int k = hlf * 32; k < hlf * 32 + 32; ++k)
    pw.ST[n * 64 + k] = (k < 16) ? (h16)pw.HOUT[k * 16 + n] : (h16)0.f;
  {
    v8f xh = forward_jnet(fd, pw.ST, pw.D1, pw.D2, lane, false);
#pragma unroll
    for (int v = 0; v < 8; ++v) outp[48 + v + hlf * 8] = xh[v];
  }

  // ===================== pass 2: x_zero -> zn =====================
  for (int k = hlf * 32; k < hlf * 32 + 32; ++k)
    pw.ST[n * 64 + k] = (k < 16) ? (h16)xp[32 + k] : (h16)0.f;
  st_network(fc, cc, HB + H_GWFC, HB + H_GWCC, pw, lane);

  for (int r = hlf * 4; r < hlf * 4 + 4; ++r) {
    float es = expf(pw.VS[r * 16 + n]);
    float sc1 = es * xp[32 + 8 + r] + 1.f;
    float sum = 0.f;
#pragma unroll
    for (int i = 0; i < 16; ++i) {
      float g = pw.TG[(r * 16 + i) * 16 + n] * sc1;
      if (i == 8 + r) g += es;                          // eye*es term in j4
      sum += g * g;
    }
    outp[32 + 8 + r] = sqrtf(sum);
  }
  if (!hlf) {
#pragma unroll
    for (int c = 0; c < 8; ++c) outp[32 + c] = 1.f;     // identity rows: norm 1
  }
}

extern "C" void kernel_launch(void* const* d_in, const int* in_sizes, int n_in,
                              void* d_out, int out_size, void* d_ws, size_t ws_size,
                              hipStream_t stream) {
  const float* x      = (const float*)d_in[0];
  const float* fcWin  = (const float*)d_in[1];
  const float* fcbin  = (const float*)d_in[2];
  const float* fcWh   = (const float*)d_in[3];
  const float* fcbh   = (const float*)d_in[4];
  const float* fcWout = (const float*)d_in[5];
  const float* fcbout = (const float*)d_in[6];
  const float* ccWin  = (const float*)d_in[7];
  const float* ccbin  = (const float*)d_in[8];
  const float* ccWh   = (const float*)d_in[9];
  const float* ccbh   = (const float*)d_in[10];
  const float* ccWout = (const float*)d_in[11];
  const float* ccbout = (const float*)d_in[12];
  const float* fdWin  = (const float*)d_in[13];
  const float* fdbin  = (const float*)d_in[14];
  const float* fdWh   = (const float*)d_in[15];
  const float* fdbh   = (const float*)d_in[16];
  const float* fdWout = (const float*)d_in[17];
  const float* fdbout = (const float*)d_in[18];
  float* out = (float*)d_out;

  int batch = in_sizes[0] / 48;
  int tiles_per_block = WAVES * NBT;
  int grid = (batch + tiles_per_block - 1) / tiles_per_block;

  diffeo_kernel<<<dim3(grid), dim3(TB), LDS_BYTES, stream>>>(
      x, fcWin, fcbin, fcWh, fcbh, fcWout, fcbout,
      ccWin, ccbin, ccWh, ccbh, ccWout, ccbout,
      fdWin, fdbin, fdWh, fdbh, fdWout, fdbout, out, batch);
}
